// Attention_39917426049170
// MI455X (gfx1250) — matrix operand order repaired and checked by reading, not compiled, not run on hardware
//
#include <hip/hip_runtime.h>
#include <hip/hip_bf16.h>

#define BATCH 8
#define SEQ   1024
#define EMB   1024
#define NH    16
#define HD    64
#define BN    (BATCH * SEQ)   // 8192

typedef __attribute__((ext_vector_type(8)))  _Float16 v8h;
typedef __attribute__((ext_vector_type(16))) _Float16 v16h;
typedef __attribute__((ext_vector_type(8)))  float    v8f;

__device__ __forceinline__ v16h cat16(v8h a, v8h b) {
    return __builtin_shufflevector(a, b, 0,1,2,3,4,5,6,7,8,9,10,11,12,13,14,15);
}
__device__ __forceinline__ v8f wmma16(v16h a, v16h b, v8f c) {
    return __builtin_amdgcn_wmma_f32_16x16x32_f16(false, a, false, b, (short)0, c, false, false);
}
__device__ __forceinline__ v8f zero8() {
    v8f z = {0.f,0.f,0.f,0.f,0.f,0.f,0.f,0.f};
    return z;
}

// ---- CDNA5 async global->LDS copy (ASYNCcnt-tracked, no VGPR data path) ----
// LDS byte address = low 32 bits of the flat address of a shared pointer.
__device__ __forceinline__ void async_b128(void* lds, const void* g) {
    unsigned l = (unsigned)(size_t)lds;
    asm volatile("global_load_async_to_lds_b128 %0, %1, off"
                 :: "v"(l), "v"((unsigned long long)(size_t)g)
                 : "memory");
}
#if __has_builtin(__builtin_amdgcn_s_wait_asynccnt)
#define WAIT_ASYNC(n) __builtin_amdgcn_s_wait_asynccnt(n)
#else
#define WAIT_ASYNC(n) asm volatile("s_wait_asynccnt " #n ::: "memory")
#endif

// ---------------------------------------------------------------- converts
__global__ void cvt_f32_to_f16(const float* __restrict__ src,
                               _Float16* __restrict__ dst, int n) {
    int i = (blockIdx.x * blockDim.x + threadIdx.x) * 4;
    int stride = gridDim.x * blockDim.x * 4;
    for (; i < n; i += stride) {
        float4 v = *(const float4*)(src + i);
        dst[i + 0] = (_Float16)v.x;
        dst[i + 1] = (_Float16)v.y;
        dst[i + 2] = (_Float16)v.z;
        dst[i + 3] = (_Float16)v.w;
    }
}

// Wt[n][k] (f16) = W[k][n] (f32); 32x32 tiles via LDS. Block = 256 threads.
__global__ __launch_bounds__(256) void transpose_cvt(
    const float* __restrict__ W, _Float16* __restrict__ Wt, int K, int N) {
    __shared__ float tile[32][33];
    int n0 = blockIdx.x * 32, k0 = blockIdx.y * 32;
    int tx = threadIdx.x & 31, ty = threadIdx.x >> 5;   // 32 x 8
    #pragma unroll
    for (int r = 0; r < 32; r += 8)
        tile[ty + r][tx] = W[(size_t)(k0 + ty + r) * N + n0 + tx];
    __syncthreads();
    #pragma unroll
    for (int r = 0; r < 32; r += 8)
        Wt[(size_t)(n0 + ty + r) * K + k0 + tx] = (_Float16)tile[tx][ty + r];
}

// ---------------------------------------------------------------- QKV GEMM
// C[8192 x 3072] = X @ Wqkv + b.  B is pre-transposed: Bt[n][k].
// Block: 256 thr = 8 waves; tile M=256 (32 rows/wave), N=64, K-step 32,
// async double-buffered LDS staging (5 async instr/wave/stage).
__global__ __launch_bounds__(256) void gemm_qkv(
    const _Float16* __restrict__ A,    // [8192][1024]
    const _Float16* __restrict__ Bt,   // [3072][1024]
    const float*    __restrict__ bias, // [3072]
    _Float16* __restrict__ qout,       // [bh][n][d]
    _Float16* __restrict__ kout,       // [bh][n][d]
    _Float16* __restrict__ vout) {     // [bh][d][n]  (transposed for attn)
    __shared__ _Float16 As[2][256 * 40];
    __shared__ _Float16 Bs[2][64 * 40];

    const int tid  = threadIdx.x;
    const int lane = tid & 31, wave = tid >> 5;
    const int n0 = blockIdx.x * 64;
    const int m0 = blockIdx.y * 256;

    v8f acc[2][4];
    for (int i = 0; i < 2; i++)
        for (int j = 0; j < 4; j++) acc[i][j] = zero8();

    #define GQ_ISSUE(buf, k0v)                                                  \
        do {                                                                    \
            _Pragma("unroll")                                                   \
            for (int j = 0; j < 4; j++) {                                       \
                int ch = j * 256 + tid;                                         \
                int row = ch >> 2, col = (ch & 3) * 8;                          \
                async_b128(&As[buf][row * 40 + col],                            \
                           A + (size_t)(m0 + row) * EMB + (k0v) + col);         \
            }                                                                   \
            {   int row = tid >> 2, col = (tid & 3) * 8;                        \
                async_b128(&Bs[buf][row * 40 + col],                            \
                           Bt + (size_t)(n0 + row) * EMB + (k0v) + col); }      \
        } while (0)

    GQ_ISSUE(0, 0);
    for (int kt = 0; kt < EMB / 32; kt++) {
        const int buf = kt & 1;
        if (kt + 1 < EMB / 32) { GQ_ISSUE(buf ^ 1, (kt + 1) * 32); WAIT_ASYNC(5); }
        else                   { WAIT_ASYNC(0); }
        __syncthreads();
        v16h bf[4];
        #pragma unroll
        for (int nt = 0; nt < 4; nt++) {
            const _Float16* brow = &Bs[buf][(nt * 16 + (lane & 15)) * 40 + (lane >> 4) * 8];
            bf[nt] = cat16(*(const v8h*)brow, *(const v8h*)(brow + 16));
        }
        #pragma unroll
        for (int ms = 0; ms < 2; ms++) {
            const _Float16* arow =
                &As[buf][(wave * 32 + ms * 16 + (lane & 15)) * 40 + (lane >> 4) * 8];
            v16h af = cat16(*(const v8h*)arow, *(const v8h*)(arow + 16));
            #pragma unroll
            for (int nt = 0; nt < 4; nt++)
                acc[ms][nt] = wmma16(af, bf[nt], acc[ms][nt]);
        }
        __syncthreads();
    }
    #undef GQ_ISSUE

    // scatter epilogue: column c -> (h, d, s) innermost-qkv factorization
    #pragma unroll
    for (int ms = 0; ms < 2; ms++) {
        const int mbase = m0 + wave * 32 + ms * 16 + (lane >> 4) * 8;
        #pragma unroll
        for (int nt = 0; nt < 4; nt++) {
            int gn = n0 + nt * 16 + (lane & 15);
            int s = gn % 3, dd = (gn / 3) % HD, h = gn / (3 * HD);
            float bs  = bias[gn];
            float scl = (s == 0) ? 0.125f : 1.0f;      // fold 1/sqrt(64) into Q
            #pragma unroll
            for (int r = 0; r < 8; r++) {
                int gm = mbase + r;
                int b = gm >> 10, n = gm & 1023;
                size_t bh = (size_t)(b * NH + h);
                float v = (acc[ms][nt][r] + bs) * scl;
                size_t idx = (s == 2) ? (bh * HD + dd) * SEQ + n
                                      : (bh * SEQ + n) * HD + dd;
                ((s == 0) ? qout : (s == 1) ? kout : vout)[idx] = (_Float16)v;
            }
        }
    }
}

// ---------------------------------------------------------------- attention
// Block = 128 thr (4 waves) per (b,h, 64-query tile). Two-pass softmax over
// 64x1032 padded f16 score LDS; K and V^T chunks async double-buffered.
__global__ __launch_bounds__(128) void attn(
    const _Float16* __restrict__ q16,   // [bh][n][d]
    const _Float16* __restrict__ k16,   // [bh][n][d]
    const _Float16* __restrict__ vt16,  // [bh][d][n]
    _Float16* __restrict__ ctx) {       // [b][n][h*64+d]
    extern __shared__ _Float16 sm[];
    _Float16* scores = sm;                        // 64 x 1032
    _Float16* qs     = sm + 64 * 1032;            // 64 x 72
    _Float16* kv0    = qs + 64 * 72;              // 64 x 72 (double buffer)
    _Float16* kv1    = kv0 + 64 * 72;
    float*    linv   = (float*)(kv1 + 64 * 72);   // 64
    _Float16* kvb[2] = {kv0, kv1};

    const int tid = threadIdx.x, lane = tid & 31, wave = tid >> 5;
    const int qb = blockIdx.x;
    const int bh = blockIdx.y;
    const _Float16* Q  = q16  + (size_t)bh * SEQ * HD;
    const _Float16* Kp = k16  + (size_t)bh * SEQ * HD;
    const _Float16* Vt = vt16 + (size_t)bh * HD * SEQ;

    // 64x64 f16 tile copy = 4 async b128 instr / wave
    #define AT_STAGE(dstp, srcp, srcld)                                        \
        do {                                                                   \
            _Pragma("unroll")                                                  \
            for (int j = 0; j < 4; j++) {                                      \
                int ch = j * 128 + tid;                                        \
                int row = ch >> 3, col = (ch & 7) * 8;                         \
                async_b128((dstp) + row * 72 + col,                            \
                           (srcp) + (size_t)row * (srcld) + col);              \
            }                                                                  \
        } while (0)

    AT_STAGE(qs, Q + (size_t)(qb * 64) * HD, HD);      // Q tile
    AT_STAGE(kv0, Kp, HD);                             // K chunk 0
    WAIT_ASYNC(4);                                     // Q done (K0 may pend)
    __syncthreads();
    const _Float16* qrow = &qs[(wave * 16 + (lane & 15)) * 72 + (lane >> 4) * 8];
    v16h qa0 = cat16(*(const v8h*)qrow,        *(const v8h*)(qrow + 16));
    v16h qa1 = cat16(*(const v8h*)(qrow + 32), *(const v8h*)(qrow + 48));

    // ---- pass 1: scores = (Q/8) K^T
    for (int kc = 0; kc < 16; kc++) {
        if (kc + 1 < 16) { AT_STAGE(kvb[(kc + 1) & 1], Kp + (size_t)((kc + 1) * 64) * HD, HD); WAIT_ASYNC(4); }
        else             { WAIT_ASYNC(0); }
        __syncthreads();
        const _Float16* kbuf = kvb[kc & 1];
        #pragma unroll
        for (int kn = 0; kn < 4; kn++) {
            const _Float16* brow = &kbuf[(kn * 16 + (lane & 15)) * 72 + (lane >> 4) * 8];
            v16h b0 = cat16(*(const v8h*)brow,        *(const v8h*)(brow + 16));
            v16h b1 = cat16(*(const v8h*)(brow + 32), *(const v8h*)(brow + 48));
            v8f s = zero8();
            s = wmma16(qa0, b0, s);
            s = wmma16(qa1, b1, s);
            int col  = kc * 64 + kn * 16 + (lane & 15);
            int mrow = wave * 16 + (lane >> 4) * 8;
            #pragma unroll
            for (int r = 0; r < 8; r++)
                scores[(mrow + r) * 1032 + col] = (_Float16)s[r];
        }
        __syncthreads();
    }

    // ---- pass 2: softmax (2 threads per row)
    {
        int row = tid >> 1;
        _Float16* sr = &scores[row * 1032 + (tid & 1) * 512];
        float mx = -1e30f;
        for (int c = 0; c < 512; c += 8) {
            v8h v = *(const v8h*)(sr + c);
            #pragma unroll
            for (int j = 0; j < 8; j++) mx = fmaxf(mx, (float)v[j]);
        }
        mx = fmaxf(mx, __shfl_xor(mx, 1, 32));
        float sum = 0.f;
        for (int c = 0; c < 512; c += 8) {
            v8h v = *(const v8h*)(sr + c);
            v8h e;
            #pragma unroll
            for (int j = 0; j < 8; j++) {
                float ev = __expf((float)v[j] - mx);
                e[j] = (_Float16)ev;
                sum += ev;
            }
            *(v8h*)(sr + c) = e;
        }
        sum += __shfl_xor(sum, 1, 32);
        if ((tid & 1) == 0) linv[row] = 1.0f / sum;
    }
    __syncthreads();

    // ---- pass 3: O = P V  (V^T tiles: rows = dfeat, cols = keys)
    v8f oacc[4];
    for (int i = 0; i < 4; i++) oacc[i] = zero8();
    AT_STAGE(kv0, Vt, SEQ);
    for (int kc = 0; kc < 16; kc++) {
        if (kc + 1 < 16) { AT_STAGE(kvb[(kc + 1) & 1], Vt + (kc + 1) * 64, SEQ); WAIT_ASYNC(4); }
        else             { WAIT_ASYNC(0); }
        __syncthreads();
        const _Float16* vbuf = kvb[kc & 1];
        const _Float16* prow =
            &scores[(wave * 16 + (lane & 15)) * 1032 + kc * 64 + (lane >> 4) * 8];
        v16h pa0 = cat16(*(const v8h*)prow,        *(const v8h*)(prow + 16));
        v16h pa1 = cat16(*(const v8h*)(prow + 32), *(const v8h*)(prow + 48));
        #pragma unroll
        for (int dn = 0; dn < 4; dn++) {
            const _Float16* brow = &vbuf[(dn * 16 + (lane & 15)) * 72 + (lane >> 4) * 8];
            v16h b0 = cat16(*(const v8h*)brow,        *(const v8h*)(brow + 16));
            v16h b1 = cat16(*(const v8h*)(brow + 32), *(const v8h*)(brow + 48));
            oacc[dn] = wmma16(pa0, b0, oacc[dn]);
            oacc[dn] = wmma16(pa1, b1, oacc[dn]);
        }
        __syncthreads();
    }
    #undef AT_STAGE

    const int b = bh >> 4, h = bh & 15;
    const int mrow = wave * 16 + (lane >> 4) * 8;
    float li[8];
    #pragma unroll
    for (int r = 0; r < 8; r++) li[r] = linv[mrow + r];
    #pragma unroll
    for (int dn = 0; dn < 4; dn++) {
        int dfeat = dn * 16 + (lane & 15);
        #pragma unroll
        for (int r = 0; r < 8; r++) {
            int n = qb * 64 + mrow + r;
            ctx[((size_t)(b * SEQ + n)) * EMB + h * HD + dfeat] =
                (_Float16)(oacc[dn][r] * li[r]);
        }
    }
}

// ---------------------------------------------------------------- out GEMM
__global__ __launch_bounds__(256) void gemm_out(
    const _Float16* __restrict__ A,    // ctx16 [8192][1024]
    const _Float16* __restrict__ Bt,   // WoutT [1024][1024]
    const float*    __restrict__ bias, // [1024]
    float* __restrict__ out) {         // [8192][1024] f32
    __shared__ _Float16 As[2][256 * 40];
    __shared__ _Float16 Bs[2][64 * 40];

    const int tid  = threadIdx.x;
    const int lane = tid & 31, wave = tid >> 5;
    const int n0 = blockIdx.x * 64;
    const int m0 = blockIdx.y * 256;

    v8f acc[2][4];
    for (int i = 0; i < 2; i++)
        for (int j = 0; j < 4; j++) acc[i][j] = zero8();

    #define GO_ISSUE(buf, k0v)                                                  \
        do {                                                                    \
            _Pragma("unroll")                                                   \
            for (int j = 0; j < 4; j++) {                                       \
                int ch = j * 256 + tid;                                         \
                int row = ch >> 2, col = (ch & 3) * 8;                          \
                async_b128(&As[buf][row * 40 + col],                            \
                           A + (size_t)(m0 + row) * EMB + (k0v) + col);         \
            }                                                                   \
            {   int row = tid >> 2, col = (tid & 3) * 8;                        \
                async_b128(&Bs[buf][row * 40 + col],                            \
                           Bt + (size_t)(n0 + row) * EMB + (k0v) + col); }      \
        } while (0)

    GO_ISSUE(0, 0);
    for (int kt = 0; kt < EMB / 32; kt++) {
        const int buf = kt & 1;
        if (kt + 1 < EMB / 32) { GO_ISSUE(buf ^ 1, (kt + 1) * 32); WAIT_ASYNC(5); }
        else                   { WAIT_ASYNC(0); }
        __syncthreads();
        v16h bf[4];
        #pragma unroll
        for (int nt = 0; nt < 4; nt++) {
            const _Float16* brow = &Bs[buf][(nt * 16 + (lane & 15)) * 40 + (lane >> 4) * 8];
            bf[nt] = cat16(*(const v8h*)brow, *(const v8h*)(brow + 16));
        }
        #pragma unroll
        for (int ms = 0; ms < 2; ms++) {
            const _Float16* arow =
                &As[buf][(wave * 32 + ms * 16 + (lane & 15)) * 40 + (lane >> 4) * 8];
            v16h af = cat16(*(const v8h*)arow, *(const v8h*)(arow + 16));
            #pragma unroll
            for (int nt = 0; nt < 4; nt++)
                acc[ms][nt] = wmma16(af, bf[nt], acc[ms][nt]);
        }
        __syncthreads();
    }
    #undef GO_ISSUE

    #pragma unroll
    for (int ms = 0; ms < 2; ms++) {
        const int mbase = m0 + wave * 32 + ms * 16 + (lane >> 4) * 8;
        #pragma unroll
        for (int nt = 0; nt < 4; nt++) {
            int gn = n0 + nt * 16 + (lane & 15);
            float bs = bias[gn];
            #pragma unroll
            for (int r = 0; r < 8; r++)
                out[(size_t)(mbase + r) * EMB + gn] = acc[ms][nt][r] + bs;
        }
    }
}

// ---------------------------------------------------------------- launcher
extern "C" void kernel_launch(void* const* d_in, const int* in_sizes, int n_in,
                              void* d_out, int out_size, void* d_ws, size_t ws_size,
                              hipStream_t stream) {
    const float* x    = (const float*)d_in[0];
    const float* Wqkv = (const float*)d_in[1];
    const float* bqkv = (const float*)d_in[2];
    const float* Wout = (const float*)d_in[3];
    const float* bout = (const float*)d_in[4];
    float* out = (float*)d_out;

    const size_t MB = 1024 * 1024;
    char* ws = (char*)d_ws;
    _Float16* x16     = (_Float16*)(ws);            // 16 MB, reused as ctx16
    _Float16* wqkvT16 = (_Float16*)(ws + 16 * MB);  //  6 MB  [3072][1024]
    _Float16* woutT16 = (_Float16*)(ws + 22 * MB);  //  2 MB  [1024][1024]
    _Float16* q16     = (_Float16*)(ws + 24 * MB);  // 16 MB  [bh][n][d]
    _Float16* k16     = (_Float16*)(ws + 40 * MB);  // 16 MB  [bh][n][d]
    _Float16* vt16    = (_Float16*)(ws + 56 * MB);  // 16 MB  [bh][d][n]
    _Float16* ctx16   = x16;

    cvt_f32_to_f16<<<2048, 256, 0, stream>>>(x, x16, BN * EMB);
    transpose_cvt<<<dim3(3 * EMB / 32, EMB / 32), 256, 0, stream>>>(Wqkv, wqkvT16, EMB, 3 * EMB);
    transpose_cvt<<<dim3(EMB / 32, EMB / 32), 256, 0, stream>>>(Wout, woutT16, EMB, EMB);

    gemm_qkv<<<dim3(3 * EMB / 64, BN / 256), 256, 0, stream>>>(
        x16, wqkvT16, bqkv, q16, k16, vt16);

    size_t lds = (size_t)(64 * 1032 + 3 * 64 * 72) * sizeof(_Float16) + 64 * sizeof(float);
    attn<<<dim3(SEQ / 64, BATCH * NH), 128, lds, stream>>>(q16, k16, vt16, ctx16);

    gemm_out<<<dim3(EMB / 64, BN / 256), 256, 0, stream>>>(ctx16, woutT16, bout, out);
}
